// HuffmanTree_24335284699569
// MI455X (gfx1250) — matrix operand order repaired and checked by reading, not compiled, not run on hardware
//
#include <hip/hip_runtime.h>
#include <math.h>

typedef __attribute__((ext_vector_type(16))) _Float16 v16h;
typedef __attribute__((ext_vector_type(8)))  _Float16 v8h;
typedef __attribute__((ext_vector_type(4)))  _Float16 v4h;
typedef __attribute__((ext_vector_type(8)))  float    v8f;
typedef __attribute__((ext_vector_type(4)))  unsigned int v4u;
typedef __attribute__((ext_vector_type(8)))  int      v8i;
typedef __attribute__((ext_vector_type(4)))  int      v4i;

// Problem constants (H fixed by the reference).
constexpr int HH      = 768;          // hidden size
constexpr int HFLOATS = 768;          // floats per half node-row (384 h * 2 outputs)
constexpr int ROWF    = 1536;         // floats per full node-row (768 h * 2 outputs)
constexpr int KCH     = 384 / 32;     // 12 K-chunks of 32 per half
constexpr int SPLITD  = 2;            // depth-split across blocks

#if defined(__has_builtin)
#if __has_builtin(__builtin_amdgcn_tensor_load_to_lds)
#define HAVE_TDM 1
#else
#define HAVE_TDM 0
#endif
#else
#define HAVE_TDM 0
#endif

#if HAVE_TDM
// One TDM gather: 16 rows (indices in g2/g3, 16-bit each) of `tileW` floats
// starting at gbase, row stride ROWF floats, into LDS at ldsOff.
__device__ __forceinline__ void tdm_gather16(const float* gbase, unsigned ldsOff,
                                             int vm1, v4i g2, v4i g3)
{
    unsigned long long ga = (unsigned long long)gbase;
    v4u g0;
    g0[0] = 0x80000001u;                                  // count=1, gather_mode=1, 16-bit idx
    g0[1] = ldsOff;                                        // lds_addr
    g0[2] = (unsigned)(ga & 0xFFFFFFFFull);                // global_addr[31:0]
    g0[3] = (unsigned)((ga >> 32) & 0x1FFFFFFull)          // global_addr[56:32]
          | 0x80000000u;                                   // type = 2 ("image")
    v8i g1;
    g1[0] = (int)(2u << 16);                               // workgroup_mask=0, data_size=4B
    g1[1] = (int)((unsigned)HFLOATS << 16);                // tensor_dim0[15:0] @ [79:64... lo]
    g1[2] = (int)(((unsigned)vm1 & 0xFFFFu) << 16);        // tensor_dim0 hi=0 | tensor_dim1 lo
    g1[3] = (int)(((unsigned)HFLOATS << 16)                // tile_dim0 = 768
          | (((unsigned)vm1 >> 16) & 0xFFFFu));            // tensor_dim1 hi
    g1[4] = 16;                                            // tile_dim1 = #valid indices
    g1[5] = ROWF;                                          // tensor_dim0_stride lo32 = 1536
    g1[6] = 0;
    g1[7] = 0;
#if __clang_major__ >= 23
    v8i z8 = {};
    __builtin_amdgcn_tensor_load_to_lds(g0, g1, g2, g3, z8, 0);
#else
    __builtin_amdgcn_tensor_load_to_lds(g0, g1, g2, g3, 0);
#endif
}

// Pack 16 per-lane node ids (lanes 0..15) into 8 dwords of 16-bit gather indices.
__device__ __forceinline__ void pack_idx(int node, v4i& g2, v4i& g3)
{
    int p[8];
#pragma unroll
    for (int q = 0; q < 8; ++q) {
        int a = __builtin_amdgcn_readlane(node, 2 * q);
        int b = __builtin_amdgcn_readlane(node, 2 * q + 1);
        p[q] = (a & 0xFFFF) | (b << 16);
    }
    g2 = (v4i){p[0], p[1], p[2], p[3]};
    g3 = (v4i){p[4], p[5], p[6], p[7]};
}
#endif // HAVE_TDM

// One wave (32 threads) per block. Block owns a 16-token tile and the depth
// subset d = dgrp, dgrp+SPLITD, ...  TDM double-buffers gathered node rows.
__global__ __launch_bounds__(32)
void hsoftmax_tdm_wmma_kernel(const float* __restrict__ hidden,
                              const int*   __restrict__ target,
                              const float* __restrict__ node_W,     // [V-1][768][2]
                              const float* __restrict__ node_b,     // [V-1][2]
                              const int*   __restrict__ path_nodes, // [V][D]
                              const int*   __restrict__ path_bits,  // [V][D]
                              const unsigned char* __restrict__ path_mask, // [V][D]
                              int T, int D, int vm1,
                              float* __restrict__ partial)
{
    __shared__ __align__(16) _Float16 hA[16 * HH];           // 24 KB f16 A tile
    __shared__ __align__(16) float    Wlds[2][16 * HFLOATS]; // 2 x 48 KB gathered W
    __shared__ float cS[2 * 8 * 32];                         // 2 KB C staging
    __shared__ int   tgtS[16];
    __shared__ float accS[32];

    const int lane = threadIdx.x;
    const int bx   = blockIdx.x;
    const int tile = bx / SPLITD;
    const int dgrp = bx % SPLITD;
    const int t0   = tile * 16;

    if (lane < 16) {
        int t = t0 + lane;
        tgtS[lane] = (t < T) ? target[t] : 0;
    }
    __syncthreads();

    // ---- stage hidden tile as f16 (float4 loads, 8B f16 stores) ----
    for (int idx = lane; idx < 16 * (HH / 4); idx += 32) {
        int row = idx / (HH / 4);
        int c4  = idx - row * (HH / 4);
        int t   = t0 + row;
        float4 v = make_float4(0.f, 0.f, 0.f, 0.f);
        if (t < T) v = ((const float4*)hidden)[(size_t)t * (HH / 4) + c4];
        v4h hv;
        hv[0] = (_Float16)v.x; hv[1] = (_Float16)v.y;
        hv[2] = (_Float16)v.z; hv[3] = (_Float16)v.w;
        *(v4h*)&hA[row * HH + c4 * 4] = hv;
    }
    __syncthreads();

    const int  col    = lane & 15;        // token slot this lane feeds (B column)
    const int  hiHalf = lane >> 4;        // 0: lanes 0-15, 1: lanes 16-31
    const bool tokV   = (t0 + col) < T;

    const int nd = (D > dgrp) ? ((D - dgrp + SPLITD - 1) / SPLITD) : 0;
    if (nd == 0) {
        if (lane == 0) partial[bx] = 0.0f;
        return;
    }

#if HAVE_TDM
    const unsigned ldsW[2] = { (unsigned)(uintptr_t)&Wlds[0][0],
                               (unsigned)(uintptr_t)&Wlds[1][0] };
    v4i g2c, g3c;
    {
        int n0 = tokV ? path_nodes[(size_t)tgtS[col] * D + dgrp] : 0;
        pack_idx(n0, g2c, g3c);
    }
    // Prologue: issue both halves of depth 0.
    tdm_gather16(node_W + 0,       ldsW[0], vm1, g2c, g3c);
    tdm_gather16(node_W + HFLOATS, ldsW[1], vm1, g2c, g3c);
#endif

    float acc = 0.0f;

    for (int k = 0; k < nd; ++k) {
        v8f c0 = {}; v8f c1 = {};

        for (int hf = 0; hf < 2; ++hf) {
            const int step = 2 * k + hf;

#if HAVE_TDM
            if (step == 2 * nd - 1) __builtin_amdgcn_s_wait_tensorcnt((short)0);
            else                    __builtin_amdgcn_s_wait_tensorcnt((short)1);
            asm volatile("" ::: "memory");   // TDM wrote LDS: no hoisting across this
#else
            // Fallback: cooperative synchronous gather of 16 half-rows.
            {
                int dk = dgrp + k * SPLITD;
                for (int idx = lane; idx < 16 * (HFLOATS / 4); idx += 32) {
                    int row = idx / (HFLOATS / 4);
                    int c4  = idx - row * (HFLOATS / 4);
                    int tg  = tgtS[row];
                    int nn  = ((t0 + row) < T) ? path_nodes[(size_t)tg * D + dk] : 0;
                    float4 w = ((const float4*)(node_W + (size_t)nn * ROWF + hf * HFLOATS))[c4];
                    ((float4*)&Wlds[step & 1][row * HFLOATS])[c4] = w;
                }
                __syncthreads();
            }
#endif
            const float* Wl = &Wlds[step & 1][0];

            for (int kc = 0; kc < KCH; ++kc) {
                // A fragment: lanes 0-15 K=base+[0..15];
                // lanes 16-31 K=base+[16..31].
                const int ha = hf * 384 + kc * 32 + hiHalf * 16;
                v8h alo = *(const v8h*)&hA[col * HH + ha];
                v8h ahi = *(const v8h*)&hA[col * HH + ha + 8];
                v16h a;
#pragma unroll
                for (int q = 0; q < 8; ++q) { a[q] = alo[q]; a[q + 8] = ahi[q]; }

                // B fragments: lanes 0-15 K=base+[0..15], lanes 16-31 K=base+[16..31].
                const int kb = kc * 32 + hiHalf * 16;
                const float* Wrow = Wl + col * HFLOATS;
                v16h b0, b1;
#pragma unroll
                for (int q = 0; q < 16; ++q) {
                    float2 w = ((const float2*)Wrow)[kb + q]; // (W[n][k][0], W[n][k][1])
                    b0[q] = (_Float16)w.x;
                    b1[q] = (_Float16)w.y;
                }

                c0 = __builtin_amdgcn_wmma_f32_16x16x32_f16(false, a, false, b0,
                                                            (short)0, c0, false, false);
                c1 = __builtin_amdgcn_wmma_f32_16x16x32_f16(false, a, false, b1,
                                                            (short)0, c1, false, false);
            }

#if HAVE_TDM
            // Re-issue over this buffer only after our LDS reads have drained.
            asm volatile("s_wait_dscnt 0x0" ::: "memory");
            const int inext = step + 2;
            if (inext < 2 * nd) {
                if ((inext & 1) == 0) {   // first half of depth k+1: new gather indices
                    int nn = tokV ? path_nodes[(size_t)tgtS[col] * D + (dgrp + (k + 1) * SPLITD)] : 0;
                    pack_idx(nn, g2c, g3c);
                }
                tdm_gather16(node_W + (inext & 1) * HFLOATS, ldsW[inext & 1], vm1, g2c, g3c);
            }
#endif
        }

        // ---- diagonal extraction via LDS (no scratch spills) ----
#pragma unroll
        for (int r = 0; r < 8; ++r) {
            cS[r * 32 + lane]       = c0[r];
            cS[256 + r * 32 + lane] = c1[r];
        }
        __syncthreads();
        if (lane < 16) {
            const int j = lane;
            const int srcLane = (j < 8) ? j : (j + 16);
            const int r = j & 7;
            float l0 = cS[r * 32 + srcLane];
            float l1 = cS[256 + r * 32 + srcLane];
            const int t = t0 + j;
            if (t < T) {
                const int dk = dgrp + k * SPLITD;
                const int tg = tgtS[j];
                const unsigned char m = path_mask[(size_t)tg * D + dk];
                if (m) {
                    const int nd2 = path_nodes[(size_t)tg * D + dk];
                    const int bit = path_bits[(size_t)tg * D + dk];
                    l0 += node_b[(size_t)nd2 * 2 + 0];
                    l1 += node_b[(size_t)nd2 * 2 + 1];
                    // probs = softmax([l0,l1]); ce = logsumexp(probs) - probs[bit]
                    float dd  = l1 - l0;
                    float p1  = 1.0f / (1.0f + expf(-dd));
                    float p0  = 1.0f - p1;
                    float lse = logf(expf(p0) + expf(p1));
                    acc += lse - (bit ? p1 : p0);
                }
            }
        }
        __syncthreads();
    }

    // Deterministic per-block reduction (fixed order).
    accS[lane] = acc;
    __syncthreads();
    if (lane == 0) {
        float s = 0.0f;
        for (int q = 0; q < 32; ++q) s += accS[q];
        partial[bx] = s;
    }
}

// Deterministic final reduction over per-block partials.
__global__ void hsoftmax_reduce_kernel(const float* __restrict__ partial, int n,
                                       float* __restrict__ out)
{
    if (threadIdx.x == 0 && blockIdx.x == 0) {
        float s = 0.0f;
        for (int i = 0; i < n; ++i) s += partial[i];
        out[0] = s;
    }
}

extern "C" void kernel_launch(void* const* d_in, const int* in_sizes, int n_in,
                              void* d_out, int out_size, void* d_ws, size_t ws_size,
                              hipStream_t stream) {
    const float*         hidden     = (const float*)d_in[0];
    const int*           target     = (const int*)d_in[1];
    const float*         node_W     = (const float*)d_in[2];
    const float*         node_b     = (const float*)d_in[3];
    const int*           path_nodes = (const int*)d_in[4];
    const int*           path_bits  = (const int*)d_in[5];
    const unsigned char* path_mask  = (const unsigned char*)d_in[6]; // jnp bool = 1 byte

    const int T   = in_sizes[1];              // B*S
    const int H   = in_sizes[0] / T;          // 768
    const int Vm1 = in_sizes[2] / (H * 2);    // V-1
    const int V   = Vm1 + 1;
    const int D   = in_sizes[4] / V;          // max path depth

    const int tiles  = (T + 15) / 16;
    const int blocks = tiles * SPLITD;
    float* partial = (float*)d_ws;            // blocks * 4 bytes scratch

    hsoftmax_tdm_wmma_kernel<<<blocks, 32, 0, stream>>>(
        hidden, target, node_W, node_b, path_nodes, path_bits, path_mask,
        T, D, Vm1, partial);
    hsoftmax_reduce_kernel<<<1, 32, 0, stream>>>(partial, blocks, (float*)d_out);
}
